// DynamicConv2d_66726611911242
// MI455X (gfx1250) — compile-verified, matrix operand order checked by reading, not run on hardware
//
#include <hip/hip_runtime.h>
#include <hip/hip_bf16.h>
#include <math.h>

typedef __bf16 bf16;
typedef __attribute__((ext_vector_type(8)))  float v8f;
typedef __attribute__((ext_vector_type(16))) bf16  v16bf;
typedef __attribute__((ext_vector_type(4)))  int   v4i;
typedef __attribute__((address_space(1))) v4i gas_v4i;  // v4i in global AS
typedef __attribute__((address_space(3))) v4i las_v4i;  // v4i in LDS AS

#define CB    32
#define CIN   64
#define COUT  128
#define HH    96
#define WW    96
#define KD    576          // CIN * 9
#define NPIX  9216         // 96*96
#define NTILE 32           // spatial positions per block
#define NBLK_P (NPIX / NTILE)   // 288
#define KCHUNKS 18         // 576 / 32

#define FRAG_BYTES 36864   // 18 chunks x 2 ntiles x 32 lanes x 32 B
#define RAW_OFF    36864   // raw x tile lives after the fragments
#define RAWC       48      // raw tile columns per (ci,row): 96 B, b128-aligned
#define RAW_BYTES  (CIN * 3 * RAWC * 2)   // 18432
#define SMEM_BYTES (FRAG_BYTES + RAW_BYTES) // 55296

#if defined(__gfx1250__) && __has_builtin(__builtin_amdgcn_global_load_async_to_lds_b128)
#define USE_ASYNC_LDS 1
#else
#define USE_ASYNC_LDS 0
#endif

// ---------------- x -> bf16 (halves HBM read traffic for the conv) ----------
__global__ __launch_bounds__(256) void cvt_bf16_kernel(const float* __restrict__ x,
                                                       bf16* __restrict__ xb, int n4) {
  int i = blockIdx.x * 256 + threadIdx.x;
  if (i >= n4) return;
  float4 f = ((const float4*)x)[i];
  union { bf16 h[4]; unsigned long long u; } u;
  u.h[0] = (bf16)f.x; u.h[1] = (bf16)f.y; u.h[2] = (bf16)f.z; u.h[3] = (bf16)f.w;
  ((unsigned long long*)xb)[i] = u.u;
}

// ---------------- per-sample stats -> cond[B,4] ------------------------------
__global__ __launch_bounds__(256) void stats_kernel(const float* __restrict__ x,
                                                    const float* __restrict__ att,
                                                    const float* __restrict__ motion,
                                                    float* __restrict__ cond) {
  __shared__ float s1[256], s2[256], s3[256];
  const int b = blockIdx.x;
  const int tid = threadIdx.x;
  const size_t xbase = (size_t)b * (CIN * NPIX);
  float sum = 0.f, sq = 0.f;
  for (int i = tid; i < CIN * NPIX; i += 256) {
    float v = x[xbase + i];
    sum += v; sq += v * v;
  }
  float as = 0.f;
  const size_t abase = (size_t)b * NPIX;
  for (int i = tid; i < NPIX; i += 256) as += att[abase + i];
  s1[tid] = sum; s2[tid] = sq; s3[tid] = as;
  __syncthreads();
  for (int off = 128; off > 0; off >>= 1) {
    if (tid < off) { s1[tid] += s1[tid+off]; s2[tid] += s2[tid+off]; s3[tid] += s3[tid+off]; }
    __syncthreads();
  }
  if (tid == 0) {
    const float N = (float)(CIN * NPIX);
    float mean = s1[0] / N;
    float var = (s2[0] - N * mean * mean) / (N - 1.0f);   // ddof=1
    var = var > 0.f ? var : 0.f;
    cond[b*4+0] = mean;
    cond[b*4+1] = sqrtf(var);
    cond[b*4+2] = 1.0f - s3[0] / (float)NPIX;
    cond[b*4+3] = motion[b];
  }
}

// ---------------- tiny MLP + softmax -> kernel_weights[B,4] ------------------
__global__ void mlp_kernel(const float* __restrict__ cond,
                           const float* __restrict__ w1, const float* __restrict__ b1,
                           const float* __restrict__ w2, const float* __restrict__ b2,
                           float* __restrict__ kwts) {
  int b = threadIdx.x;
  if (b >= CB) return;
  float c[4];
  for (int j = 0; j < 4; j++) c[j] = cond[b*4 + j];
  float h[16];
  for (int i = 0; i < 16; i++) {
    float a = b1[i];
    for (int j = 0; j < 4; j++) a += c[j] * w1[j*16 + i];
    h[i] = a > 0.f ? a : 0.f;
  }
  float lg[4]; float mx = -1e30f;
  for (int k = 0; k < 4; k++) {
    float a = b2[k];
    for (int i = 0; i < 16; i++) a += h[i] * w2[i*4 + k];
    lg[k] = a; mx = a > mx ? a : mx;
  }
  float s = 0.f;
  for (int k = 0; k < 4; k++) { lg[k] = __expf(lg[k] - mx); s += lg[k]; }
  for (int k = 0; k < 4; k++) kwts[b*4 + k] = lg[k] / s;
}

// ---------------- mix base kernels -> per-sample bf16 weights [B,128,576] ----
__global__ __launch_bounds__(256) void dynw_kernel(const float* __restrict__ kwts,
                                                   const float* __restrict__ basek,
                                                   bf16* __restrict__ dynW) {
  int idx = blockIdx.x * 256 + threadIdx.x;
  if (idx >= CB * COUT * KD) return;
  int b   = idx / (COUT * KD);
  int rem = idx - b * (COUT * KD);
  int o   = rem / KD;
  int kk  = rem - o * KD;            // kk = ci*9 + (kh*3+kw)
  float acc = 0.f;
  #pragma unroll
  for (int k = 0; k < 4; k++)
    acc += kwts[b*4 + k] * basek[(size_t)(k * COUT + o) * KD + kk];
  dynW[(size_t)idx] = (bf16)acc;     // [b][o][kk] row-major
}

// ---------------- implicit-GEMM conv via v_wmma_f32_16x16x32_bf16 ------------
// Block: 256 threads = 8 waves. Wave w computes rows M = 16w..16w+15 for 32
// spatial columns (two 16x16 N-tiles).
// LDS: [0, 36864)       im2col B-fragments in native WMMA register layout
//      [36864, 55296)   raw x tile: 64 ci x 3 rows x 48 cols bf16
union ABu { v16bf v; float4 f4[2]; };

__global__ __launch_bounds__(256) void conv_wmma_kernel(const bf16* __restrict__ xbf,
                                                        const bf16* __restrict__ dynW,
                                                        const float* __restrict__ bias,
                                                        float* __restrict__ out) {
  __shared__ __align__(32) unsigned char smem[SMEM_BYTES];
  const int blk = blockIdx.x;
  const int b   = blk / NBLK_P;
  const int p0  = (blk - b * NBLK_P) * NTILE;   // 32-wide, never crosses a row
  const int tid = threadIdx.x;

  const int y  = p0 / WW;                 // uniform over the whole block
  const int x0 = p0 - y * WW;             // tile's first column in the row
  const int colbase = (x0 * 3) >> 2;      // {0,24,48}: covers halo cols, 16B-aligned
  const size_t xsamp = (size_t)b * CIN * NPIX;

  // ---- stage 1: raw x tile -> LDS via async 128-bit copies -----------------
  // 64 ci x 3 rows x 6 segments of 8 bf16 = 1152 lane transfers per block.
  for (int e = tid; e < CIN * 3 * (RAWC / 8); e += 256) {
    int seg = e % 6;
    int rc  = e / 6;                      // ci*3 + row
    int row = rc % 3;
    int ci  = rc / 3;
    int iy  = y + row - 1;
    if ((unsigned)iy < (unsigned)HH) {
      const bf16* gsrc = xbf + xsamp + ((size_t)ci * HH + iy) * WW + colbase + seg * 8;
      unsigned char* ldst = smem + RAW_OFF + ((rc * RAWC + seg * 8) << 1);
#if USE_ASYNC_LDS
      __builtin_amdgcn_global_load_async_to_lds_b128(
          (gas_v4i*)gsrc, (las_v4i*)ldst, 0, 0);
#else
      *(float4*)(void*)ldst = *(const float4*)(const void*)gsrc;
#endif
    }
  }
#if USE_ASYNC_LDS
#if __has_builtin(__builtin_amdgcn_s_wait_asynccnt)
  __builtin_amdgcn_s_wait_asynccnt(0);
#else
  asm volatile("s_wait_asynccnt 0" ::: "memory");
#endif
#endif
  __syncthreads();

  // ---- stage 2: build im2col B-fragments from raw LDS tile -----------------
  // B (32x16, bf16): lane n<16 holds col N=n, K = {0..7, 16..23}; lane n+16
  // holds K = {8..15, 24..31}; groups of 4 consecutive K are contiguous and
  // 8B-aligned in a fragment half -> one ds_store_b64 per group.
  for (int g = tid; g < (KD * NTILE) / 4; g += 256) {
    int n  = g & 31;                      // column in tile (== tid&31)
    int gk = g >> 5;                      // K-group index (0..143), K=4*gk
    int kc   = gk >> 3;                   // 32-wide K chunk
    int k0   = (gk << 2) & 31;            // K offset inside chunk
    int khi  = k0 >> 3;
    int klo0 = k0 & 7;                    // 0 or 4
    int t    = n >> 4, nl = n & 15;
    int lane = nl | ((khi & 1) << 4);
    int half = khi >> 1;
    int off  = (((kc * 2 + t) * 32 + lane) << 5) + (half << 4) + (klo0 << 1);
    int kgb = gk << 2;                    // global K of first element
    int ci  = kgb / 9;
    int r   = kgb - ci * 9;               // tap index kh*3+kw
    int xx  = x0 + n;
    union { bf16 h[4]; unsigned long long u; } u;
    #pragma unroll
    for (int i = 0; i < 4; ++i) {
      int kh = r / 3;
      int kw = r - kh * 3;
      int iy = y + kh - 1, ix = xx + kw - 1;
      bf16 v = (bf16)0.0f;
      if ((unsigned)iy < (unsigned)HH && (unsigned)ix < (unsigned)WW)
        v = *(const bf16*)(smem + RAW_OFF +
                           (((ci * 3 + kh) * RAWC + (ix - colbase)) << 1));
      u.h[i] = v;
      if (++r == 9) { r = 0; ++ci; }      // advance K incrementally
    }
    *(unsigned long long*)(smem + off) = u.u;
  }
  __syncthreads();

  // ---- stage 3: WMMA main loop ---------------------------------------------
  const int wave = tid >> 5, lane = tid & 31;
  const int m0   = wave << 4;
  const int mrow = m0 + (lane & 15);
  const int hi   = lane >> 4;                 // K-half select for A/B layout

  v8f c0 = {}; v8f c1 = {};
  // A fragment base: lane holds row `mrow`, K = {base+hi*8 .. +7, base+16+hi*8 ..}
  const bf16* wbase = dynW + ((size_t)b * COUT + mrow) * KD + hi * 8;
  __builtin_prefetch(wbase, 0, 1);            // global_prefetch_b8
  const unsigned char* smlane = smem + (lane << 5);

  for (int kc = 0; kc < KCHUNKS; ++kc) {
    ABu a;
    a.f4[0] = *(const float4*)(const void*)(wbase + kc * 32);        // K +0..7
    a.f4[1] = *(const float4*)(const void*)(wbase + kc * 32 + 16);   // K +16..23
    const v16bf bf0 = *(const v16bf*)(const void*)(smlane + ((kc * 2 + 0) << 10));
    const v16bf bf1 = *(const v16bf*)(const void*)(smlane + ((kc * 2 + 1) << 10));
    c0 = __builtin_amdgcn_wmma_f32_16x16x32_bf16(false, a.v, false, bf0,
                                                 (short)0, c0, false, false);
    c1 = __builtin_amdgcn_wmma_f32_16x16x32_bf16(false, a.v, false, bf1,
                                                 (short)0, c1, false, false);
  }

  // ---- scatter accumulators: VGPR r, lane<16 -> M=r, lane>=16 -> M=r+8 ----
  const int nlo = lane & 15;
  const int mb  = hi * 8;
  #pragma unroll
  for (int r = 0; r < 8; ++r) {
    int m = m0 + mb + r;
    float bv = bias[m];
    size_t rowoff = ((size_t)b * COUT + m) * NPIX + p0;
    out[rowoff + nlo]      = c0[r] + bv;
    out[rowoff + 16 + nlo] = c1[r] + bv;
  }
}

// ---------------- host-side orchestration -----------------------------------
extern "C" void kernel_launch(void* const* d_in, const int* in_sizes, int n_in,
                              void* d_out, int out_size, void* d_ws, size_t ws_size,
                              hipStream_t stream) {
  const float* x      = (const float*)d_in[0];
  const float* att    = (const float*)d_in[1];
  const float* motion = (const float*)d_in[2];
  const float* basek  = (const float*)d_in[3];
  const float* w1     = (const float*)d_in[4];
  const float* b1     = (const float*)d_in[5];
  const float* w2     = (const float*)d_in[6];
  const float* b2     = (const float*)d_in[7];
  const float* bias   = (const float*)d_in[8];
  float* out = (float*)d_out;

  char* ws = (char*)d_ws;
  // workspace layout (all 256-B aligned):
  //   [0)                xbf  : 32*64*96*96 bf16 = 37,748,736 B
  //   [37748736)         dynW : 32*128*576 bf16  =  4,718,592 B
  //   [42467328)         cond : 32*4 f32
  //   [42467840)         kwts : 32*4 f32
  bf16*  xbf  = (bf16*)(ws);
  bf16*  dynW = (bf16*)(ws + 37748736);
  float* cond = (float*)(ws + 42467328);
  float* kwts = (float*)(ws + 42467840);

  const int nx4 = (CB * CIN * NPIX) / 4;                 // 4,718,592
  cvt_bf16_kernel<<<(nx4 + 255) / 256, 256, 0, stream>>>(x, xbf, nx4);

  stats_kernel<<<CB, 256, 0, stream>>>(x, att, motion, cond);

  mlp_kernel<<<1, 32, 0, stream>>>(cond, w1, b1, w2, b2, kwts);

  const int ndw = CB * COUT * KD;                        // 2,359,296
  dynw_kernel<<<(ndw + 255) / 256, 256, 0, stream>>>(kwts, basek, dynW);

  conv_wmma_kernel<<<CB * NBLK_P, 256, 0, stream>>>(xbf, dynW, bias, out);
}
